// QuantizeSignedActPercentage_66838281060426
// MI455X (gfx1250) — compile-verified, matrix operand order checked by reading, not run on hardware
//
#include <hip/hip_runtime.h>
#include <cstdint>

typedef __attribute__((ext_vector_type(16))) _Float16 v16h;
typedef __attribute__((ext_vector_type(8)))  float    v8f;
typedef __attribute__((ext_vector_type(4)))  unsigned v4u;
typedef __attribute__((ext_vector_type(8)))  int      v8i;
typedef __attribute__((ext_vector_type(4)))  int      v4i;

#define HW        3136u            // 56*56
#define NCHAN     256
#define CH_STRIDE 802816u          // 256*3136 (stride between batch-chunks of a channel)
#define CH_ELEMS  100352u          // 32*3136 elements per channel
#define THREADS   1024u
#define REG_V4    6                // 6 float4 per thread in VGPRs (24 floats, 96KB/WG)
#define LDS_V4    18               // 18 float4 per thread in LDS (+1 float2 tail)
#define LDS_ELEMS 75776u           // 74 floats * 1024 threads
#define LDS_BYTES ((size_t)LDS_ELEMS * 4)        // 303104
// LDS region holds channel elements [24576, 100352), channel-linear:
//   24576 = 7*3136 + 2624  -> (a) 512 elems of chunk n=7, (b) chunks n=8..31 (24 rows x 3136)

#if defined(__HIP_DEVICE_COMPILE__) && __has_builtin(__builtin_amdgcn_tensor_load_to_lds) && __has_builtin(__builtin_amdgcn_s_wait_tensorcnt)
#define USE_TDM 1
#else
#define USE_TDM 0
#endif

#if USE_TDM
// Build D# per CDNA5 ISA ch.8 and issue TENSOR_LOAD_TO_LDS (uniform args -> SGPRs).
// This toolchain exposes the 6-arg builtin: (v4u g0, v8i g1, v4i g2, v4i g3, v8i gx, i32 cpol).
__device__ __forceinline__ void tdm_load(unsigned lds_byte_off, unsigned long long gaddr,
                                         unsigned dim0, unsigned dim1,
                                         unsigned tile0, unsigned tile1,
                                         unsigned long long stride0_elems) {
  v4u g0;
  g0[0] = 1u;                                            // count=1 (valid), user mode
  g0[1] = lds_byte_off;                                  // lds_addr [63:32]
  g0[2] = (unsigned)gaddr;                               // global_addr lo
  g0[3] = (unsigned)((gaddr >> 32) & 0x1FFFFFFull) | 0x80000000u; // addr hi | type=2
  v8i g1;
  g1[0] = (int)(2u << 16);                               // data_size=2 (4 bytes); no mask/flags
  g1[1] = (int)((dim0 & 0xFFFFu) << 16);                 // tensor_dim0 lo16 (bits 63:48)
  g1[2] = (int)((dim0 >> 16) | ((dim1 & 0xFFFFu) << 16));// dim0 hi | tensor_dim1 lo
  g1[3] = (int)((dim1 >> 16) | (tile0 << 16));           // dim1 hi | tile_dim0
  g1[4] = (int)(tile1);                                  // tile_dim1 (tile_dim2=0)
  g1[5] = (int)((unsigned)stride0_elems);                // tensor_dim0_stride lo32
  g1[6] = (int)((unsigned)((stride0_elems >> 32) & 0xFFFFull)); // stride hi16 (dim1_stride=0)
  g1[7] = 0;
  v4i g2 = {0, 0, 0, 0};
  v4i g3 = {0, 0, 0, 0};
  v8i gx = {0, 0, 0, 0, 0, 0, 0, 0};
  __builtin_amdgcn_tensor_load_to_lds(g0, g1, g2, g3, gx, 0);
}
#endif

__global__ __launch_bounds__(1024)
void quant_percentile_kernel(const float* __restrict__ x, float* __restrict__ out) {
  const int      c     = blockIdx.x;            // one workgroup per channel
  const unsigned t     = threadIdx.x;
  const int      lane  = (int)(t & 31u);
  const int      wid   = (int)(t >> 5);
  const unsigned cbase = (unsigned)c * HW;

  extern __shared__ float sdata[];              // channel elements [24576,100352), linear
  __shared__ float red[2][128];                 // per-wave count partials (double buffered)
  __shared__ float wred[64];                    // min/max wave partials
  __shared__ float bc[2];                       // broadcast min/max

  const float4* sd4 = (const float4*)sdata;     // 18944 float4s
  const float2* sd2 = (const float2*)sdata;     // tail pairs at index 36864+t

  // ---- stage 1: first 24576 channel elems -> VGPRs, vectorized b128, coalesced ----
  float4 rv[REG_V4];
#pragma unroll
  for (int q = 0; q < REG_V4; ++q) {
    unsigned j = 4u * ((unsigned)q * THREADS + t);        // never crosses a 3136-chunk
    unsigned n = j / HW, i = j - n * HW;
    rv[q] = *(const float4*)(x + (size_t)n * CH_STRIDE + cbase + i);
  }

  // ---- stage 1b: bulk of channel -> LDS via Tensor Data Mover ----
#if USE_TDM
  if (t == 0u) {
    unsigned lb = (unsigned)(unsigned long long)(uintptr_t)(void*)sdata;
    unsigned long long base = (unsigned long long)(uintptr_t)x;
    // (a) 1-D: 512 elems, chunk n=7 starting at i=2624
    unsigned long long ga = base + 4ull * ((unsigned long long)7u * CH_STRIDE + cbase + 2624u);
    tdm_load(lb, ga, 512u, 1u, 512u, 0u, 512ull);
    // (b) 2-D: 24 rows x 3136, row stride = CH_STRIDE elements
    unsigned long long gb = base + 4ull * ((unsigned long long)8u * CH_STRIDE + cbase);
    tdm_load(lb + 512u * 4u, gb, 3136u, 24u, 3136u, 24u, (unsigned long long)CH_STRIDE);
    __builtin_amdgcn_s_wait_tensorcnt((short)0);
  }
  __syncthreads();
#else
#pragma unroll
  for (int q = 0; q < LDS_V4; ++q) {
    unsigned j = 24576u + 4u * ((unsigned)q * THREADS + t);
    unsigned n = j / HW, i = j - n * HW;
    ((float4*)sdata)[(unsigned)q * THREADS + t] = *(const float4*)(x + (size_t)n * CH_STRIDE + cbase + i);
  }
  {
    unsigned j = 98304u + 2u * t;                          // all in chunk n=31
    ((float2*)sdata)[36864u + t] = *(const float2*)(x + (size_t)31u * CH_STRIDE + cbase + (j - 31u * HW));
  }
  __syncthreads();
#endif

  // ---- stage 2: per-channel min/max ----
  float mn = 3.402823466e38f, mx = -3.402823466e38f;
#pragma unroll
  for (int q = 0; q < REG_V4; ++q) {
    mn = fminf(fminf(fminf(fminf(mn, rv[q].x), rv[q].y), rv[q].z), rv[q].w);
    mx = fmaxf(fmaxf(fmaxf(fmaxf(mx, rv[q].x), rv[q].y), rv[q].z), rv[q].w);
  }
#pragma unroll
  for (int q = 0; q < LDS_V4; ++q) {
    float4 v = sd4[(unsigned)q * THREADS + t];
    mn = fminf(fminf(fminf(fminf(mn, v.x), v.y), v.z), v.w);
    mx = fmaxf(fmaxf(fmaxf(fmaxf(mx, v.x), v.y), v.z), v.w);
  }
  {
    float2 v = sd2[36864u + t];
    mn = fminf(fminf(mn, v.x), v.y);
    mx = fmaxf(fmaxf(mx, v.x), v.y);
  }
#pragma unroll
  for (int off = 16; off > 0; off >>= 1) {
    mn = fminf(mn, __shfl_down(mn, (unsigned)off));
    mx = fmaxf(mx, __shfl_down(mx, (unsigned)off));
  }
  if (lane == 0) { wred[wid] = mn; wred[32 + wid] = mx; }
  __syncthreads();
  if (wid == 0) {
    float a = wred[lane], b = wred[32 + lane];
#pragma unroll
    for (int off = 16; off > 0; off >>= 1) {
      a = fminf(a, __shfl_down(a, (unsigned)off));
      b = fmaxf(b, __shfl_down(b, (unsigned)off));
    }
    if (lane == 0) { bc[0] = a; bc[1] = b; }
  }
  __syncthreads();
  mn = bc[0]; mx = bc[1];

  // ---- stage 3: dual binary search (upper tail & lower tail) ----
  float lu = mn, ru = mx;   // upper: frac(x > pivot) in [P0,P1]
  float ll = mn, rl = mx;   // lower: frac(x < pivot) in [P0,P1]
#pragma unroll 1
  for (int it = 0; it < 50; ++it) {
    const int   buf = it & 1;
    const float pu  = 0.5f * (lu + ru);
    const float pl  = 0.5f * (ll + rl);

    // wave-level counts via ballot/popcount (SALU accumulate co-issues with VALU cmp).
    // Parts: a = 24 reg + 36 LDS (60/lane, <=1920), b = 36 LDS + 2 tail (38/lane, <=1216):
    // both <= 2048 so the f16 WMMA partials are exact.
    int cua = 0, cub = 0, cla = 0, clb = 0;
#pragma unroll
    for (int q = 0; q < REG_V4; ++q) {
      cua += __popcll(__ballot(rv[q].x > pu)) + __popcll(__ballot(rv[q].y > pu))
           + __popcll(__ballot(rv[q].z > pu)) + __popcll(__ballot(rv[q].w > pu));
      cla += __popcll(__ballot(rv[q].x < pl)) + __popcll(__ballot(rv[q].y < pl))
           + __popcll(__ballot(rv[q].z < pl)) + __popcll(__ballot(rv[q].w < pl));
    }
#pragma unroll
    for (int q = 0; q < 9; ++q) {
      float4 v = sd4[(unsigned)q * THREADS + t];
      cua += __popcll(__ballot(v.x > pu)) + __popcll(__ballot(v.y > pu))
           + __popcll(__ballot(v.z > pu)) + __popcll(__ballot(v.w > pu));
      cla += __popcll(__ballot(v.x < pl)) + __popcll(__ballot(v.y < pl))
           + __popcll(__ballot(v.z < pl)) + __popcll(__ballot(v.w < pl));
    }
#pragma unroll
    for (int q = 9; q < LDS_V4; ++q) {
      float4 v = sd4[(unsigned)q * THREADS + t];
      cub += __popcll(__ballot(v.x > pu)) + __popcll(__ballot(v.y > pu))
           + __popcll(__ballot(v.z > pu)) + __popcll(__ballot(v.w > pu));
      clb += __popcll(__ballot(v.x < pl)) + __popcll(__ballot(v.y < pl))
           + __popcll(__ballot(v.z < pl)) + __popcll(__ballot(v.w < pl));
    }
    {
      float2 v = sd2[36864u + t];
      cub += __popcll(__ballot(v.x > pu)) + __popcll(__ballot(v.y > pu));
      clb += __popcll(__ballot(v.x < pl)) + __popcll(__ballot(v.y < pl));
    }
    if (lane == 0) {
      red[buf][wid * 4 + 0] = (float)cua;
      red[buf][wid * 4 + 1] = (float)cub;
      red[buf][wid * 4 + 2] = (float)cla;
      red[buf][wid * 4 + 3] = (float)clb;
    }
    __syncthreads();

    // Cross-wave reduction of the 128 partials with one v_wmma_f32_16x16x32_f16:
    // A rows 0..3 = {cu_a, cu_b, cl_a, cl_b} over K=wave 0..31, B = ones -> D rows = totals.
    float Cu, Cl;
#if __has_builtin(__builtin_amdgcn_wmma_f32_16x16x32_f16)
    {
      const int mrow = lane & 15;
      const int half = lane >> 4;
      v16h A, B;
#pragma unroll
      for (int e = 0; e < 16; ++e) {
        int K = e + 8 * half + ((e >= 8) ? 8 : 0);      // 16-bit A 16x32 lane/VGPR layout
        float val = red[buf][K * 4 + (mrow & 3)];
        A[e] = (_Float16)((mrow < 4) ? val : 0.0f);
        B[e] = (_Float16)1.0f;
      }
      v8f Cz;
#pragma unroll
      for (int e = 0; e < 8; ++e) Cz[e] = 0.0f;
      v8f D = __builtin_amdgcn_wmma_f32_16x16x32_f16(false, A, false, B, (short)0, Cz,
                                                     false, false);
      Cu = D[0] + D[1];   // valid on lanes 0..15 (rows 0/1, any column)
      Cl = D[2] + D[3];   // rows 2/3
      Cu = __shfl(Cu, 0);
      Cl = __shfl(Cl, 0);
    }
#else
    {
      Cu = 0.0f; Cl = 0.0f;
      for (int wv = 0; wv < 32; ++wv) {
        Cu += red[buf][wv * 4 + 0] + red[buf][wv * 4 + 1];
        Cl += red[buf][wv * 4 + 2] + red[buf][wv * 4 + 3];
      }
    }
#endif
    const float fu = Cu / (float)CH_ELEMS;   // matches jnp.mean: exact int sum, f32 divide
    const float fl = Cl / (float)CH_ELEMS;

    bool moved = false;
    if (fu > 0.02f) { lu = pu; moved = true; }
    if (fu < 0.01f) { ru = pu; moved = true; }
    if (fl < 0.01f) { ll = pl; moved = true; }
    if (fl > 0.02f) { rl = pl; moved = true; }
    if (!moved) break;   // both searches converged; remaining reference iters are no-ops
  }

  // ---- stage 4: quantize + store (single HBM write, b128/b64 coalesced) ----
  const float upper = 0.5f * (lu + ru);
  const float lower = 0.5f * (ll + rl);
  const float bias  = 0.5f * (lower + upper);
  const float shift = 8.0f - log2f(upper - lower);
  const float rb    = rintf(bias);                       // round half-to-even, as jnp.round
  const float sc    = ldexpf(1.0f, (int)rintf(shift));   // exact exp2 of integer

#define QZ(v) fminf(fmaxf(floorf(((v) - rb) * sc), -128.0f), 127.0f)
#pragma unroll
  for (int q = 0; q < REG_V4; ++q) {
    unsigned j = 4u * ((unsigned)q * THREADS + t);
    unsigned n = j / HW, i = j - n * HW;
    float4 o;
    o.x = QZ(rv[q].x); o.y = QZ(rv[q].y); o.z = QZ(rv[q].z); o.w = QZ(rv[q].w);
    *(float4*)(out + (size_t)n * CH_STRIDE + cbase + i) = o;
  }
#pragma unroll
  for (int q = 0; q < LDS_V4; ++q) {
    unsigned j = 24576u + 4u * ((unsigned)q * THREADS + t);
    unsigned n = j / HW, i = j - n * HW;
    float4 v = sd4[(unsigned)q * THREADS + t];
    float4 o;
    o.x = QZ(v.x); o.y = QZ(v.y); o.z = QZ(v.z); o.w = QZ(v.w);
    *(float4*)(out + (size_t)n * CH_STRIDE + cbase + i) = o;
  }
  {
    unsigned j = 98304u + 2u * t;                          // all in chunk n=31
    float2 v = sd2[36864u + t];
    float2 o;
    o.x = QZ(v.x); o.y = QZ(v.y);
    *(float2*)(out + (size_t)31u * CH_STRIDE + cbase + (j - 31u * HW)) = o;
  }
#undef QZ
}

extern "C" void kernel_launch(void* const* d_in, const int* in_sizes, int n_in,
                              void* d_out, int out_size, void* d_ws, size_t ws_size,
                              hipStream_t stream) {
  (void)in_sizes; (void)n_in; (void)out_size; (void)d_ws; (void)ws_size;
  const float* x  = (const float*)d_in[0];
  float* out      = (float*)d_out;
  hipLaunchKernelGGL(quant_percentile_kernel, dim3(NCHAN), dim3(THREADS), LDS_BYTES, stream,
                     x, out);
}
